// ChangeGuideModule_4750233829930
// MI455X (gfx1250) — matrix operand order repaired and checked by reading, not compiled, not run on hardware
//
#include <hip/hip_runtime.h>
#include <hip/hip_bf16.h>
#include <math.h>

typedef __attribute__((ext_vector_type(16))) __bf16 v16bf;
typedef __attribute__((ext_vector_type(8)))  __bf16 v8bf;
typedef __attribute__((ext_vector_type(8)))  float  v8f;
typedef __attribute__((ext_vector_type(4)))  float  v4f;
typedef int v4i __attribute__((__vector_size__(4 * sizeof(int))));

#define BATCH 4
#define CDIM  256
#define CQ    32
#define NPIX  4096   // 64*64

#if defined(__AMDGCN__) && __has_builtin(__builtin_amdgcn_global_load_async_to_lds_b128)
#define USE_ASYNC 1
#define AS1 __attribute__((address_space(1)))
#define AS3 __attribute__((address_space(3)))
#else
#define USE_ASYNC 0
#endif

__device__ __forceinline__ void wait_async0() {
#if defined(__AMDGCN__)
#if __has_builtin(__builtin_amdgcn_s_wait_asynccnt)
    __builtin_amdgcn_s_wait_asynccnt(0);
#else
    asm volatile("s_wait_asynccnt 0x0" ::: "memory");
#endif
    asm volatile("" ::: "memory");   // compiler-level barrier: force LDS reloads
#endif
}

__device__ __forceinline__ v16bf cat8(v8bf lo, v8bf hi) {
    return __builtin_shufflevector(lo, hi, 0,1,2,3,4,5,6,7,8,9,10,11,12,13,14,15);
}

// ---------------------------------------------------------------------------
// Kernel 1: gate = 1 + sigmoid(bilinear_align_corners(g, 32->64))
// ---------------------------------------------------------------------------
__global__ void gate_kernel(const float* __restrict__ g, float* __restrict__ gate) {
    int idx = blockIdx.x * blockDim.x + threadIdx.x;
    if (idx >= BATCH * NPIX) return;
    int b = idx >> 12;
    int p = idx & 4095;
    int y = p >> 6, x = p & 63;
    const float s = 31.0f / 63.0f;
    float fy = y * s, fx = x * s;
    int y0 = (int)floorf(fy); y0 = min(max(y0, 0), 31);
    int x0 = (int)floorf(fx); x0 = min(max(x0, 0), 31);
    int y1 = min(y0 + 1, 31), x1 = min(x0 + 1, 31);
    float wy = fy - (float)y0, wx = fx - (float)x0;
    const float* gb = g + b * 1024;
    float g00 = gb[y0 * 32 + x0], g01 = gb[y0 * 32 + x1];
    float g10 = gb[y1 * 32 + x0], g11 = gb[y1 * 32 + x1];
    float top = g00 * (1.f - wx) + g01 * wx;
    float bot = g10 * (1.f - wx) + g11 * wx;
    float v   = top * (1.f - wy) + bot * wy;
    gate[idx] = 1.0f + 1.0f / (1.0f + __expf(-v));
}

// ---------------------------------------------------------------------------
// Kernel 2: projection  out = (W[O,256] @ x[b,:,n] + bias) * gate   (bf16 out)
// Block: 128 threads (4 waves). Grid: (N/16, ceil(O/64), B).
// transpose_out=1: out[b,n,o]  (A-/B-operand rows for attention: qT, kT)
// transpose_out=0: out[b,o,n]  (PV B-operand rows: v)
// ---------------------------------------------------------------------------
#define XSTR 264   // padded c-stride (264*2 = 528 B, 16B aligned rows)
__global__ void proj_kernel(const float* __restrict__ x,
                            const float* __restrict__ Wt,
                            const float* __restrict__ bias,
                            const float* __restrict__ gate,
                            __bf16* __restrict__ out,
                            int O, int transpose_out) {
    __shared__ __bf16 X_lds[16][XSTR];   // [n][c] tile
    const int b   = blockIdx.z;
    const int n0  = blockIdx.x * 16;
    const int tid = threadIdx.x;
    const int wave = tid >> 5, lane = tid & 31;
    const int half = lane >> 4, lm = lane & 15;

    for (int i = tid; i < CDIM * 16; i += 128) {
        int c = i >> 4, n = i & 15;
        X_lds[n][c] = (__bf16)x[((size_t)b * CDIM + c) * NPIX + n0 + n];
    }
    __syncthreads();

    const int o0 = blockIdx.y * 64 + wave * 16;
    if (o0 >= O) return;

    const float* wrow = Wt + (size_t)(o0 + lm) * CDIM;
    v8f acc = {};
#pragma unroll
    for (int kk = 0; kk < 8; ++kk) {
        // A (W): rows = o, K = c.  elements 0..7 -> K=kk*32+8*half+0..7,
        //                           elements 8..15 -> +16.  Wide f32 loads + cvt.
        v4f w0 = *(const v4f*)(wrow + kk * 32 + 8 * half);
        v4f w1 = *(const v4f*)(wrow + kk * 32 + 8 * half + 4);
        v4f w2 = *(const v4f*)(wrow + kk * 32 + 16 + 8 * half);
        v4f w3 = *(const v4f*)(wrow + kk * 32 + 16 + 8 * half + 4);
        v16bf a;
#pragma unroll
        for (int e = 0; e < 4; ++e) {
            a[e]      = (__bf16)w0[e];
            a[e + 4]  = (__bf16)w1[e];
            a[e + 8]  = (__bf16)w2[e];
            a[e + 12] = (__bf16)w3[e];
        }
        // B (x): lane = n, elements = K = c in A's order (8*half, 16+8*half) -> two ds b128 loads
        v8bf blo = *(const v8bf*)&X_lds[lm][kk * 32 + 8 * half];
        v8bf bhi = *(const v8bf*)&X_lds[lm][kk * 32 + 16 + 8 * half];
        acc = __builtin_amdgcn_wmma_f32_16x16x32_bf16(false, a, false, cat8(blo, bhi),
                                                      (short)0, acc, false, false);
    }

    const int n = n0 + lm;
    const float gv = gate[b * NPIX + n];
    if (transpose_out) {
        // lane writes 8 consecutive o -> one 16B store
        v8bf ov;
#pragma unroll
        for (int r = 0; r < 8; ++r)
            ov[r] = (__bf16)((acc[r] + bias[o0 + r + 8 * half]) * gv);
        *(v8bf*)(out + ((size_t)b * NPIX + n) * O + o0 + 8 * half) = ov;
    } else {
#pragma unroll
        for (int r = 0; r < 8; ++r) {
            int o = o0 + r + 8 * half;
            out[((size_t)b * O + o) * NPIX + n] = (__bf16)((acc[r] + bias[o]) * gv);
        }
    }
}

// ---------------------------------------------------------------------------
// Kernel 3: flash attention + fused epilogue  out = gamma*(P@V^T)/rowsum + x
// Block: 128 threads (4 waves). Grid: (N/16, B). 16 query rows per block.
// V tile (64n x 256c, 36KB) staged to LDS with async-to-LDS b128 copies that
// overlap the QK^T WMMA + softmax phase; drained with s_wait_asynccnt.
// ---------------------------------------------------------------------------
#define VSTR 72   // padded n-stride (144 B rows: 16B aligned, conflict-free)
__global__ void attn_kernel(const __bf16* __restrict__ qT,   // [B,N,32]
                            const __bf16* __restrict__ kT,   // [B,N,32]
                            const __bf16* __restrict__ vM,   // [B,256,N]
                            const float*  __restrict__ xin,  // [B,256,N]
                            const float*  __restrict__ gamma,
                            float* __restrict__ out) {
    __shared__ float  S_lds[16][65];
    __shared__ __bf16 P_lds[16][VSTR];
    __shared__ __bf16 V_lds[CDIM][VSTR];   // [c][n-local], 36 KB
    __shared__ float  redmax[16][8];
    __shared__ float  redsum[16][8];
    __shared__ float  rowmax[16], rowsum[16], rowscale[16];

    const int b   = blockIdx.y;
    const int m0  = blockIdx.x * 16;
    const int tid = threadIdx.x;
    const int wave = tid >> 5, lane = tid & 31;
    const int half = lane >> 4, lm = lane & 15;

    // Q fragment (A-operand, Cq == 32 == one K-step): two b128 loads
    const __bf16* qrow = qT + ((size_t)b * NPIX + m0 + lm) * CQ;
    const v16bf aq = cat8(*(const v8bf*)(qrow + 8 * half),
                          *(const v8bf*)(qrow + 16 + 8 * half));

    v8f acc[4] = {};
    if (tid < 16) { rowmax[tid] = -INFINITY; rowsum[tid] = 0.0f; }
    __syncthreads();

    const __bf16* vbase = vM + (size_t)b * CDIM * NPIX;

    for (int n0 = 0; n0 < NPIX; n0 += 64) {
        // ---- stage V tile: wave w copies the 64 c-rows its PV consumes ----
#pragma unroll 4
        for (int j = 0; j < 16; ++j) {
            int chunk = lane + 32 * j;            // 0..511 within wave's slice
            int row   = wave * 64 + (chunk >> 3); // c row
            int nq    = chunk & 7;                // 16B chunk within 64 n
            const __bf16* gp = vbase + (size_t)row * NPIX + n0 + nq * 8;
            __bf16* lp = &V_lds[row][nq * 8];
#if USE_ASYNC
            __builtin_amdgcn_global_load_async_to_lds_b128(
                (AS1 v4i*)(unsigned long long)gp, (AS3 v4i*)lp, 0, 0);
#else
            *(v8bf*)lp = *(const v8bf*)gp;
#endif
        }

        // ---- S = Q K^T for this wave's 16-column slice (overlaps copy) ----
        const __bf16* krow = kT + ((size_t)b * NPIX + n0 + wave * 16 + lm) * CQ;
        v16bf bk = cat8(*(const v8bf*)(krow + 8 * half),
                        *(const v8bf*)(krow + 16 + 8 * half));
        v8f s = {};
        s = __builtin_amdgcn_wmma_f32_16x16x32_bf16(false, aq, false, bk,
                                                    (short)0, s, false, false);
#pragma unroll
        for (int r = 0; r < 8; ++r)
            S_lds[r + 8 * half][wave * 16 + lm] = s[r];
        __syncthreads();

        // ---- online softmax: per-row max over the 64-col tile ----
        {
            int row = tid >> 3, ch = tid & 7;
            float mx = -INFINITY;
#pragma unroll
            for (int j = 0; j < 8; ++j) mx = fmaxf(mx, S_lds[row][ch * 8 + j]);
            redmax[row][ch] = mx;
        }
        __syncthreads();
        if (tid < 16) {
            float mx = rowmax[tid];
#pragma unroll
            for (int j = 0; j < 8; ++j) mx = fmaxf(mx, redmax[tid][j]);
            rowscale[tid] = __expf(rowmax[tid] - mx);
            rowmax[tid]   = mx;
        }
        __syncthreads();

        // ---- P = exp(S - max) -> bf16 LDS; rescale accumulators ----
        {
            int row = tid >> 3, ch = tid & 7;
            float mx = rowmax[row], sum = 0.0f;
#pragma unroll
            for (int j = 0; j < 8; ++j) {
                float p = __expf(S_lds[row][ch * 8 + j] - mx);
                P_lds[row][ch * 8 + j] = (__bf16)p;
                sum += p;
            }
            redsum[row][ch] = sum;
        }
#pragma unroll
        for (int r = 0; r < 8; ++r) {
            float sc = rowscale[r + 8 * half];
            acc[0][r] *= sc; acc[1][r] *= sc; acc[2][r] *= sc; acc[3][r] *= sc;
        }
        __syncthreads();
        if (tid < 16) {
            float sm = 0.0f;
#pragma unroll
            for (int j = 0; j < 8; ++j) sm += redsum[tid][j];
            rowsum[tid] = rowsum[tid] * rowscale[tid] + sm;
        }

        // ---- drain this wave's async V copies, then O += P @ V^T ----
        wait_async0();
#pragma unroll
        for (int kk = 0; kk < 2; ++kk) {
            v16bf ap = cat8(*(const v8bf*)&P_lds[lm][kk * 32 + 8 * half],
                            *(const v8bf*)&P_lds[lm][kk * 32 + 16 + 8 * half]);
#pragma unroll
            for (int cc = 0; cc < 4; ++cc) {
                int row = wave * 64 + cc * 16 + lm;
                v16bf bv = cat8(*(const v8bf*)&V_lds[row][kk * 32 + 8 * half],
                                *(const v8bf*)&V_lds[row][kk * 32 + 16 + 8 * half]);
                acc[cc] = __builtin_amdgcn_wmma_f32_16x16x32_bf16(
                    false, ap, false, bv, (short)0, acc[cc], false, false);
            }
        }
        __syncthreads();
    }

    // ---- epilogue: normalize, gamma*out + x ; m-contiguous b128 I/O ----
    const float g = gamma[0];
    float inv[8];
#pragma unroll
    for (int r = 0; r < 8; ++r) inv[r] = 1.0f / rowsum[r + 8 * half];
#pragma unroll
    for (int cc = 0; cc < 4; ++cc) {
        int c = wave * 64 + cc * 16 + lm;
        size_t base = ((size_t)b * CDIM + c) * NPIX + m0 + 8 * half;
        v4f x0 = *(const v4f*)(xin + base);
        v4f x1 = *(const v4f*)(xin + base + 4);
        v4f o0v, o1v;
#pragma unroll
        for (int r = 0; r < 4; ++r) {
            o0v[r] = g * acc[cc][r]     * inv[r]     + x0[r];
            o1v[r] = g * acc[cc][r + 4] * inv[r + 4] + x1[r];
        }
        *(v4f*)(out + base)     = o0v;
        *(v4f*)(out + base + 4) = o1v;
    }
}

// ---------------------------------------------------------------------------
extern "C" void kernel_launch(void* const* d_in, const int* in_sizes, int n_in,
                              void* d_out, int out_size, void* d_ws, size_t ws_size,
                              hipStream_t stream) {
    const float* x  = (const float*)d_in[0];
    const float* gm = (const float*)d_in[1];
    const float* Wq = (const float*)d_in[2];
    const float* bq = (const float*)d_in[3];
    const float* Wk = (const float*)d_in[4];
    const float* bk = (const float*)d_in[5];
    const float* Wv = (const float*)d_in[6];
    const float* bv = (const float*)d_in[7];
    const float* gamma = (const float*)d_in[8];
    float* out = (float*)d_out;

    // workspace layout (all 16B-aligned offsets)
    char* ws = (char*)d_ws;
    float*  gate = (float*)ws;   ws += (size_t)BATCH * NPIX * sizeof(float);
    __bf16* qT   = (__bf16*)ws;  ws += (size_t)BATCH * NPIX * CQ * sizeof(__bf16);
    __bf16* kT   = (__bf16*)ws;  ws += (size_t)BATCH * NPIX * CQ * sizeof(__bf16);
    __bf16* vM   = (__bf16*)ws;

    gate_kernel<<<(BATCH * NPIX + 255) / 256, 256, 0, stream>>>(gm, gate);

    // q^T, k^T: [B,N,32] (K-contiguous rows for QK A/B operands)
    proj_kernel<<<dim3(NPIX / 16, 1, BATCH), 128, 0, stream>>>(x, Wq, bq, gate, qT, CQ, 1);
    proj_kernel<<<dim3(NPIX / 16, 1, BATCH), 128, 0, stream>>>(x, Wk, bk, gate, kT, CQ, 1);
    // v: [B,256,N] (n-contiguous rows for PV B-operand)
    proj_kernel<<<dim3(NPIX / 16, 4, BATCH), 128, 0, stream>>>(x, Wv, bv, gate, vM, CDIM, 0);

    attn_kernel<<<dim3(NPIX / 16, BATCH), 128, 0, stream>>>(qT, kT, vM, x, gamma, out);
}
